// RoadNet_17197049053735
// MI455X (gfx1250) — matrix operand order repaired and checked by reading, not compiled, not run on hardware
//
#include <hip/hip_runtime.h>
#include <hip/hip_bf16.h>
#include <math.h>

#define B_ 2
#define N_ 1024
#define D_ 256
#define H_ 8
#define HD_ 32
#define RH_ 64

typedef _Float16 v16h __attribute__((ext_vector_type(16)));
typedef float    v8f  __attribute__((ext_vector_type(8)));

union V16HU { v16h v; float4 f4[2]; };

__device__ __forceinline__ v8f wmma_f16(v16h a, v16h b, v8f c) {
  return __builtin_amdgcn_wmma_f32_16x16x32_f16(false, a, false, b, (short)0, c, false, false);
}

// A-matrix 16x32 f16 layout (ISA 7.12.2): lane group (lane>=16) selects K-chunk.
// per lane: elems 0-7 = K = (lane/16)*8 .. +7 ; elems 8-15 = K = 16 + (lane/16)*8 .. +7
__device__ __forceinline__ v16h load_a16(const _Float16* rowbase, int hsel) {
  V16HU u;
  u.f4[0] = *(const float4*)(rowbase + hsel * 8);
  u.f4[1] = *(const float4*)(rowbase + 16 + hsel * 8);
  return u.v;
}
// B-matrix 32x16 f16 layout: lane holds col (lane%16), elems 0-7 = K = (lane/16)*8 .. +7 ; elems 8-15 = K = 16 + (lane/16)*8 .. +7
__device__ __forceinline__ v16h load_b16(const _Float16* colbase, int hsel) {
  V16HU u;
  u.f4[0] = *(const float4*)(colbase + hsel * 8);
  u.f4[1] = *(const float4*)(colbase + 16 + hsel * 8);
  return u.v;
}
__device__ __forceinline__ v8f vzero8() {
  v8f z;
#pragma unroll
  for (int q = 0; q < 8; q++) z[q] = 0.f;
  return z;
}

// LDS byte address of a __shared__ object: generic->LDS mapping truncates to addr[31:0]
__device__ __forceinline__ unsigned lds_off(const void* p) {
  return (unsigned)(size_t)p;
}
// gfx1250 async copy global -> LDS, 16 bytes per lane, tracked by ASYNCcnt
__device__ __forceinline__ void async_load_b128(unsigned lds_dst, const void* gaddr) {
  asm volatile("global_load_async_to_lds_b128 %0, %1, off"
               :: "v"(lds_dst), "v"((unsigned long long)(size_t)gaddr)
               : "memory");
}
__device__ __forceinline__ void wait_async0() {
  asm volatile("s_wait_asynccnt 0x0" ::: "memory");
}

// ---------------- K0: per-node geometry prep ----------------
__global__ void k_prep(const float* __restrict__ rv, float* __restrict__ nodegeo) {
  int n = blockIdx.x * blockDim.x + threadIdx.x;
  if (n >= B_ * N_) return;
  const float* r = rv + (size_t)n * 9;
  float sx = r[0], sy = r[1];
  float h = atan2f(r[3] - sy, r[2] - sx);
  float4 o; o.x = sx; o.y = sy; o.z = cosf(h); o.w = sinf(h);
  ((float4*)nodegeo)[n] = o;
}

// ---------------- K1: input MLP (wave per row) ----------------
__global__ void k_inmlp(const float* __restrict__ rv,
                        const float* __restrict__ W1, const float* __restrict__ b1,
                        const float* __restrict__ g1, const float* __restrict__ be1,
                        const float* __restrict__ W2, const float* __restrict__ b2,
                        const float* __restrict__ gn, const float* __restrict__ ben,
                        float* __restrict__ xout, _Float16* __restrict__ xnout) {
  __shared__ float sh[8][D_];
  int wave = threadIdx.x >> 5, lane = threadIdx.x & 31;
  int row = blockIdx.x * 8 + wave;
  const float* r = rv + (size_t)row * 9;
  float rk[9];
#pragma unroll
  for (int k = 0; k < 9; k++) rk[k] = r[k];
  float h[8];
#pragma unroll
  for (int u = 0; u < 8; u++) {
    int j = lane + 32 * u;
    float acc = b1[j];
#pragma unroll
    for (int k = 0; k < 9; k++) acc += rk[k] * W1[k * D_ + j];
    h[u] = acc;
  }
  float s = 0.f, sq = 0.f;
#pragma unroll
  for (int u = 0; u < 8; u++) { s += h[u]; sq += h[u] * h[u]; }
  for (int m = 1; m < 32; m <<= 1) { s += __shfl_xor(s, m, 32); sq += __shfl_xor(sq, m, 32); }
  float mu = s * (1.f / D_), var = sq * (1.f / D_) - mu * mu;
  float rstd = rsqrtf(var + 1e-5f);
#pragma unroll
  for (int u = 0; u < 8; u++) {
    int j = lane + 32 * u;
    float v = (h[u] - mu) * rstd * g1[j] + be1[j];
    sh[wave][j] = fmaxf(v, 0.f);
  }
  asm volatile("s_wait_dscnt 0x0" ::: "memory");
  float x[8];
#pragma unroll
  for (int u = 0; u < 8; u++) x[u] = b2[lane + 32 * u];
  for (int k = 0; k < D_; k++) {
    float hv = sh[wave][k];
    const float* w = W2 + (size_t)k * D_;
#pragma unroll
    for (int u = 0; u < 8; u++) x[u] += hv * w[lane + 32 * u];
  }
  s = 0.f; sq = 0.f;
#pragma unroll
  for (int u = 0; u < 8; u++) { s += x[u]; sq += x[u] * x[u]; }
  for (int m = 1; m < 32; m <<= 1) { s += __shfl_xor(s, m, 32); sq += __shfl_xor(sq, m, 32); }
  mu = s * (1.f / D_); var = sq * (1.f / D_) - mu * mu;
  rstd = rsqrtf(var + 1e-5f);
#pragma unroll
  for (int u = 0; u < 8; u++) {
    int j = lane + 32 * u;
    xout[(size_t)row * D_ + j] = x[u];
    float v = (x[u] - mu) * rstd * gn[j] + ben[j];
    xnout[(size_t)row * D_ + j] = (_Float16)v;
  }
}

// ---------------- K2: RPE pair features + 5->64->8 MLP via WMMA ----------------
__global__ void k_rpe(const float* __restrict__ nodegeo,
                      const float* __restrict__ Wr1, const float* __restrict__ br1,
                      const float* __restrict__ gr, const float* __restrict__ ber,
                      const float* __restrict__ Wr2, const float* __restrict__ br2,
                      _Float16* __restrict__ bias_out) {
  __shared__ __attribute__((aligned(32))) _Float16 ldsW1[RH_][32];   // BT[n][k], k>=5 zero
  __shared__ __attribute__((aligned(32))) _Float16 ldsW2[16][RH_];   // BT[n][k], n>=8 zero
  __shared__ __attribute__((aligned(32))) _Float16 ldsH[8][16][RH_]; // per-wave hidden
  int t = threadIdx.x;
#pragma unroll
  for (int c = 0; c < 8; c++) { // 64*32
    int idx = t + 256 * c, n = idx >> 5, k = idx & 31;
    ldsW1[n][k] = (k < 5) ? (_Float16)Wr1[k * RH_ + n] : (_Float16)0.f;
  }
#pragma unroll
  for (int c = 0; c < 4; c++) { // 16*64
    int idx = t + 256 * c, n = idx >> 6, k = idx & 63;
    ldsW2[n][k] = (n < 8) ? (_Float16)Wr2[k * H_ + n] : (_Float16)0.f;
  }
  __syncthreads();

  int wave = t >> 5, lane = t & 31;
  int blk = blockIdx.x;                 // B*N*8
  int jb = blk & 7, i = (blk >> 3) & (N_ - 1), b = blk >> 13;
  int j0 = jb * 128 + wave * 16;

  float4 gi = ((const float4*)nodegeo)[b * N_ + i];
  float4 gj = ((const float4*)nodegeo)[b * N_ + j0 + (lane & 15)];
  v16h a;
#pragma unroll
  for (int q = 0; q < 16; q++) a[q] = (_Float16)0.f;
  if (lane < 16) { // K=0..7 chunk lives in lanes 0-15; feats occupy K=0..4
    float dx = gi.x - gj.x, dy = gi.y - gj.y;
    float dist = sqrtf(dx * dx + dy * dy);
    a[0] = (_Float16)logf(dist + 1e-5f);
    a[1] = (_Float16)(gi.z * gj.z + gi.w * gj.w);   // cos(hi-hj)
    a[2] = (_Float16)(gi.w * gj.z - gi.z * gj.w);   // sin(hi-hj)
    a[3] = (_Float16)(dx * gi.z + dy * gi.w);       // rx
    a[4] = (_Float16)(-dx * gi.w + dy * gi.z);      // ry
  }

  v8f acc[4];
#pragma unroll
  for (int tt = 0; tt < 4; tt++) {
    float bb = br1[tt * 16 + (lane & 15)];
#pragma unroll
    for (int q = 0; q < 8; q++) acc[tt][q] = bb;
    v16h bm = load_b16(&ldsW1[tt * 16 + (lane & 15)][0], lane >> 4);
    acc[tt] = wmma_f16(a, bm, acc[tt]);
  }
  // LayerNorm over 64 per pair-row (C layout: row=q+8*(lane>>4), col striped over 16 lanes)
  float mu[8], rs[8];
#pragma unroll
  for (int q = 0; q < 8; q++) {
    float s = acc[0][q] + acc[1][q] + acc[2][q] + acc[3][q];
    float sq = acc[0][q] * acc[0][q] + acc[1][q] * acc[1][q] +
               acc[2][q] * acc[2][q] + acc[3][q] * acc[3][q];
    for (int m = 1; m < 16; m <<= 1) { s += __shfl_xor(s, m, 32); sq += __shfl_xor(sq, m, 32); }
    float m_ = s * (1.f / RH_), v_ = sq * (1.f / RH_) - m_ * m_;
    mu[q] = m_; rs[q] = rsqrtf(v_ + 1e-5f);
  }
#pragma unroll
  for (int tt = 0; tt < 4; tt++) {
    int col = tt * 16 + (lane & 15);
    float g = gr[col], be = ber[col];
#pragma unroll
    for (int q = 0; q < 8; q++) {
      int rowm = q + 8 * (lane >> 4);
      float v = (acc[tt][q] - mu[q]) * rs[q] * g + be;
      ldsH[wave][rowm][col] = (_Float16)fmaxf(v, 0.f);
    }
  }
  asm volatile("s_wait_dscnt 0x0" ::: "memory");

  v8f acc2;
  {
    int col = lane & 15;
    float bb = (col < 8) ? br2[col] : 0.f;
#pragma unroll
    for (int q = 0; q < 8; q++) acc2[q] = bb;
  }
#pragma unroll
  for (int s2 = 0; s2 < 2; s2++) {
    v16h a2 = load_a16(&ldsH[wave][lane & 15][s2 * 32], lane >> 4);
    v16h b2 = load_b16(&ldsW2[lane & 15][s2 * 32], lane >> 4);
    acc2 = wmma_f16(a2, b2, acc2);
  }
  int hh = lane & 15;
  if (hh < 8) {
    _Float16* dst = bias_out + (((size_t)b * H_ + hh) * N_ + i) * N_;
#pragma unroll
    for (int q = 0; q < 8; q++) dst[j0 + q + 8 * (lane >> 4)] = (_Float16)acc2[q];
  }
}

// ---------------- K3: QKV GEMM (2048x768x256), async A staging ----------------
__global__ void k_qkv(const _Float16* __restrict__ xn,
                      const float* __restrict__ Wqkv, const float* __restrict__ bqkv,
                      _Float16* __restrict__ qo, _Float16* __restrict__ ko,
                      _Float16* __restrict__ vto) {
  __shared__ __attribute__((aligned(32))) _Float16 ldsA[128][32];
  __shared__ __attribute__((aligned(32))) _Float16 ldsBT[64][32];
  int t = threadIdx.x, wave = t >> 5, lane = t & 31;
  int mb = blockIdx.x & 15, nb = blockIdx.x >> 4;
  int row0 = mb * 128, col0 = nb * 64;
  v8f acc[4];
#pragma unroll
  for (int tt = 0; tt < 4; tt++) {
    float bb = bqkv[col0 + tt * 16 + (lane & 15)];
#pragma unroll
    for (int q = 0; q < 8; q++) acc[tt][q] = bb;
  }
  for (int ks = 0; ks < D_; ks += 32) {
#pragma unroll
    for (int c = 0; c < 2; c++) { // async copy A tile: 128x32 f16
      int idx = t + 256 * c, rr = idx >> 2, ko2 = (idx & 3) * 8;
      async_load_b128(lds_off(&ldsA[rr][ko2]),
                      &xn[(size_t)(row0 + rr) * D_ + ks + ko2]);
    }
#pragma unroll
    for (int c = 0; c < 8; c++) { // f32->f16 weight conversion via VGPRs
      int idx = t + 256 * c, kk = idx >> 6, nn = idx & 63;
      ldsBT[nn][kk] = (_Float16)Wqkv[(size_t)(ks + kk) * 768 + col0 + nn];
    }
    wait_async0();
    __syncthreads();
    v16h aA = load_a16(&ldsA[wave * 16 + (lane & 15)][0], lane >> 4);
#pragma unroll
    for (int tt = 0; tt < 4; tt++) {
      v16h bB = load_b16(&ldsBT[tt * 16 + (lane & 15)][0], lane >> 4);
      acc[tt] = wmma_f16(aA, bB, acc[tt]);
    }
    __syncthreads();
  }
  const float qscale = 0.17677669529663687f; // 1/sqrt(32)
#pragma unroll
  for (int tt = 0; tt < 4; tt++) {
    int cg = col0 + tt * 16 + (lane & 15);
    int which = cg >> 8, hh = (cg >> 5) & 7, hd = cg & 31;
    float sc = (which == 0) ? qscale : 1.f;
#pragma unroll
    for (int q = 0; q < 8; q++) {
      int rowg = row0 + wave * 16 + q + 8 * (lane >> 4);
      int bb2 = rowg >> 10, nn2 = rowg & (N_ - 1);
      _Float16 val = (_Float16)(acc[tt][q] * sc);
      if (which == 2) // V stored transposed: (B,H,HD,N) for async tile staging
        vto[(((size_t)bb2 * H_ + hh) * HD_ + hd) * N_ + nn2] = val;
      else if (which == 1)
        ko[(((size_t)bb2 * H_ + hh) * N_ + nn2) * HD_ + hd] = val;
      else
        qo[(((size_t)bb2 * H_ + hh) * N_ + nn2) * HD_ + hd] = val;
    }
  }
}

// ---------------- K4: attention (flash-style, async double-buffered V) ----------------
__global__ void k_attn(const _Float16* __restrict__ qm, const _Float16* __restrict__ km,
                       const _Float16* __restrict__ vtm, const _Float16* __restrict__ biasm,
                       const unsigned char* __restrict__ mask,
                       _Float16* __restrict__ attn_out) {
  __shared__ __attribute__((aligned(32))) _Float16 ldsVT[2][32][32]; // [buf][hd][j]
  __shared__ __attribute__((aligned(32))) _Float16 ldsP[8][16][32];
  int t = threadIdx.x, wave = t >> 5, lane = t & 31;
  int blk = blockIdx.x;                 // B*H*8
  int itile = blk & 7, hh = (blk >> 3) & 7, b = blk >> 6;
  int i0 = itile * 128 + wave * 16;
  size_t headoff = ((size_t)b * H_ + hh) * N_;
  size_t vthead = ((size_t)b * H_ + hh) * HD_;

  v16h aQ = load_a16(&qm[(headoff + i0 + (lane & 15)) * HD_], lane >> 4);
  float mrow[8], lsum[8];
  v8f accO[2];
#pragma unroll
  for (int q = 0; q < 8; q++) { mrow[q] = -1e30f; lsum[q] = 0.f; }
  accO[0] = vzero8(); accO[1] = vzero8();

  // stage V tile (32 hd rows x 32 j cols) buf via async b128: threads 0..127, 16B each
  auto stage = [&](int buf, int j0) {
    if (t < 128) {
      int hd = t >> 2, part = t & 3;
      async_load_b128(lds_off(&ldsVT[buf][hd][part * 8]),
                      &vtm[(vthead + hd) * N_ + j0 + part * 8]);
    }
  };

  stage(0, 0);
  wait_async0();
  __syncthreads();
  int buf = 0;

  for (int j0 = 0; j0 < N_; j0 += 32) {
    if (j0 + 32 < N_) stage(buf ^ 1, j0 + 32); // prefetch next tile, overlapped
    __builtin_prefetch(&biasm[(headoff + i0) * N_ + j0 + 32], 0, 1);

    v8f s[2];
#pragma unroll
    for (int nt = 0; nt < 2; nt++) {
      v16h bK = load_b16(&km[(headoff + j0 + nt * 16 + (lane & 15)) * HD_], lane >> 4);
      s[nt] = wmma_f16(aQ, bK, vzero8());
    }
#pragma unroll
    for (int nt = 0; nt < 2; nt++) {
      int jc = j0 + nt * 16 + (lane & 15);
      float mf = mask[b * N_ + jc] ? 0.f : -1e30f;
#pragma unroll
      for (int q = 0; q < 8; q++) {
        int ir = i0 + q + 8 * (lane >> 4);
        s[nt][q] += (float)biasm[(headoff + ir) * N_ + jc] + mf;
      }
    }
#pragma unroll
    for (int q = 0; q < 8; q++) {
      float tm = fmaxf(s[0][q], s[1][q]);
      for (int m = 1; m < 16; m <<= 1) tm = fmaxf(tm, __shfl_xor(tm, m, 32));
      float mn = fmaxf(mrow[q], tm);
      float scale = __expf(mrow[q] - mn);
      float p0 = __expf(s[0][q] - mn), p1 = __expf(s[1][q] - mn);
      s[0][q] = p0; s[1][q] = p1;
      float ps = p0 + p1;
      for (int m = 1; m < 16; m <<= 1) ps += __shfl_xor(ps, m, 32);
      lsum[q] = lsum[q] * scale + ps;
      accO[0][q] *= scale; accO[1][q] *= scale;
      mrow[q] = mn;
    }
#pragma unroll
    for (int nt = 0; nt < 2; nt++)
#pragma unroll
      for (int q = 0; q < 8; q++)
        ldsP[wave][q + 8 * (lane >> 4)][nt * 16 + (lane & 15)] = (_Float16)s[nt][q];
    asm volatile("s_wait_dscnt 0x0" ::: "memory");
    v16h aP = load_a16(&ldsP[wave][lane & 15][0], lane >> 4);
#pragma unroll
    for (int nt = 0; nt < 2; nt++) {
      v16h bV = load_b16(&ldsVT[buf][nt * 16 + (lane & 15)][0], lane >> 4);
      accO[nt] = wmma_f16(aP, bV, accO[nt]);
    }
    wait_async0();
    __syncthreads();
    buf ^= 1;
  }
#pragma unroll
  for (int nt = 0; nt < 2; nt++) {
    int hd = nt * 16 + (lane & 15);
#pragma unroll
    for (int q = 0; q < 8; q++) {
      int ir = i0 + q + 8 * (lane >> 4);
      float v = (lsum[q] > 0.f) ? accO[nt][q] / lsum[q] : 0.f;
      attn_out[((size_t)b * N_ + ir) * D_ + hh * HD_ + hd] = (_Float16)v;
    }
  }
}

// ---------------- K5: output projection + residual + mask ----------------
__global__ void k_oproj(const _Float16* __restrict__ attn,
                        const float* __restrict__ Wo, const float* __restrict__ bo,
                        const float* __restrict__ x, const unsigned char* __restrict__ mask,
                        float* __restrict__ out) {
  __shared__ __attribute__((aligned(32))) _Float16 ldsA[128][32];
  __shared__ __attribute__((aligned(32))) _Float16 ldsBT[64][32];
  int t = threadIdx.x, wave = t >> 5, lane = t & 31;
  int mb = blockIdx.x & 15, nb = blockIdx.x >> 4;
  int row0 = mb * 128, col0 = nb * 64;
  v8f acc[4];
#pragma unroll
  for (int tt = 0; tt < 4; tt++) {
    float bb = bo[col0 + tt * 16 + (lane & 15)];
#pragma unroll
    for (int q = 0; q < 8; q++) acc[tt][q] = bb;
  }
  for (int ks = 0; ks < D_; ks += 32) {
#pragma unroll
    for (int c = 0; c < 2; c++) {
      int idx = t + 256 * c, rr = idx >> 2, ko2 = (idx & 3) * 8;
      async_load_b128(lds_off(&ldsA[rr][ko2]),
                      &attn[(size_t)(row0 + rr) * D_ + ks + ko2]);
    }
#pragma unroll
    for (int c = 0; c < 8; c++) {
      int idx = t + 256 * c, kk = idx >> 6, nn = idx & 63;
      ldsBT[nn][kk] = (_Float16)Wo[(size_t)(ks + kk) * D_ + col0 + nn];
    }
    wait_async0();
    __syncthreads();
    v16h aA = load_a16(&ldsA[wave * 16 + (lane & 15)][0], lane >> 4);
#pragma unroll
    for (int tt = 0; tt < 4; tt++) {
      v16h bB = load_b16(&ldsBT[tt * 16 + (lane & 15)][0], lane >> 4);
      acc[tt] = wmma_f16(aA, bB, acc[tt]);
    }
    __syncthreads();
  }
#pragma unroll
  for (int tt = 0; tt < 4; tt++) {
    int cg = col0 + tt * 16 + (lane & 15);
#pragma unroll
    for (int q = 0; q < 8; q++) {
      int rowg = row0 + wave * 16 + q + 8 * (lane >> 4);
      int bb2 = rowg >> 10, nn2 = rowg & (N_ - 1);
      float res = x[(size_t)rowg * D_ + cg] + acc[tt][q];
      res *= mask[bb2 * N_ + nn2] ? 1.f : 0.f;
      out[(size_t)rowg * D_ + cg] = res;
    }
  }
}

extern "C" void kernel_launch(void* const* d_in, const int* in_sizes, int n_in,
                              void* d_out, int out_size, void* d_ws, size_t ws_size,
                              hipStream_t stream) {
  const float* rv   = (const float*)d_in[0];
  const unsigned char* mask = (const unsigned char*)d_in[1];
  const float* W1   = (const float*)d_in[2];
  const float* b1   = (const float*)d_in[3];
  const float* g1   = (const float*)d_in[4];
  const float* be1  = (const float*)d_in[5];
  const float* W2   = (const float*)d_in[6];
  const float* b2   = (const float*)d_in[7];
  const float* Wr1  = (const float*)d_in[8];
  const float* br1  = (const float*)d_in[9];
  const float* gr   = (const float*)d_in[10];
  const float* ber  = (const float*)d_in[11];
  const float* Wr2  = (const float*)d_in[12];
  const float* br2  = (const float*)d_in[13];
  const float* Wqkv = (const float*)d_in[14];
  const float* bqkv = (const float*)d_in[15];
  const float* Wo   = (const float*)d_in[16];
  const float* bo   = (const float*)d_in[17];
  const float* gn   = (const float*)d_in[18];
  const float* ben  = (const float*)d_in[19];
  float* out = (float*)d_out;

  char* w = (char*)d_ws;
  auto take = [&](size_t bytes) -> char* {
    char* p = w; w += (bytes + 255) & ~(size_t)255; return p;
  };
  float*     nodegeo = (float*)take((size_t)B_ * N_ * 4 * sizeof(float));
  float*     xbuf    = (float*)take((size_t)B_ * N_ * D_ * sizeof(float));
  _Float16*  xn      = (_Float16*)take((size_t)B_ * N_ * D_ * sizeof(_Float16));
  _Float16*  qb      = (_Float16*)take((size_t)B_ * H_ * N_ * HD_ * sizeof(_Float16));
  _Float16*  kb      = (_Float16*)take((size_t)B_ * H_ * N_ * HD_ * sizeof(_Float16));
  _Float16*  vtb     = (_Float16*)take((size_t)B_ * H_ * N_ * HD_ * sizeof(_Float16));
  _Float16*  biasb   = (_Float16*)take((size_t)B_ * H_ * N_ * N_ * sizeof(_Float16));
  _Float16*  attnb   = (_Float16*)take((size_t)B_ * N_ * D_ * sizeof(_Float16));

  k_prep<<<(B_ * N_ + 255) / 256, 256, 0, stream>>>(rv, nodegeo);
  k_inmlp<<<B_ * N_ / 8, 256, 0, stream>>>(rv, W1, b1, g1, be1, W2, b2, gn, ben, xbuf, xn);
  k_rpe<<<B_ * N_ * 8, 256, 0, stream>>>(nodegeo, Wr1, br1, gr, ber, Wr2, br2, biasb);
  k_qkv<<<16 * 12, 256, 0, stream>>>(xn, Wqkv, bqkv, qb, kb, vtb);
  k_attn<<<B_ * H_ * 8, 256, 0, stream>>>(qb, kb, vtb, biasb, mask, attnb);
  k_oproj<<<16 * 4, 256, 0, stream>>>(attnb, Wo, bo, xbuf, mask, out);
}
